// MultiHeadAttention_20143396619176
// MI455X (gfx1250) — matrix operand order repaired and checked by reading, not compiled, not run on hardware
//
#include <hip/hip_runtime.h>

typedef __attribute__((ext_vector_type(16))) _Float16 v16h;
typedef __attribute__((ext_vector_type(8)))  _Float16 v8h;
typedef __attribute__((ext_vector_type(8)))  float    v8f;
typedef __attribute__((ext_vector_type(4)))  float    v4f;
typedef __attribute__((ext_vector_type(4)))  unsigned int u32x4;
typedef __attribute__((ext_vector_type(8)))  int      i32x8;
typedef __attribute__((ext_vector_type(4)))  int      i32x4;

#define B_  4
#define S_  2048
#define D_  1024
#define H_  16
#define HD_ 64

// ---------------------------------------------------------------------------
// CDNA5 async copy: global -> LDS, 16B per lane, tracked by ASYNCcnt.
// ---------------------------------------------------------------------------
static __device__ __forceinline__ void async_ld_b128(unsigned lds_off, const void* g) {
  asm volatile("global_load_async_to_lds_b128 %0, %1, off"
               :: "v"(lds_off), "v"((unsigned long long)(size_t)g)
               : "memory");
}
static __device__ __forceinline__ void wait_async0() {
  asm volatile("s_wait_asynccnt 0" ::: "memory");
}

// ---------------------------------------------------------------------------
// TDM: 2D tile load (128 rows x 32 halves) from a row-major f16 tensor with
// stride 1024 halves into LDS, padded 16 DWORDs data + 4 DWORDs pad
// (= 40-half LDS row stride).  D# layout per CDNA5 ISA.
// This toolchain exposes the 6-arg builtin (g0, g1, g2, g3, g4, cpol).
// ---------------------------------------------------------------------------
static __device__ __forceinline__ void tdm_load_tile_128x32(unsigned lds_off,
                                                            const _Float16* gptr) {
  const unsigned long long ga = (unsigned long long)(size_t)gptr;
  u32x4 g0;
  g0[0] = 1u;                                   // count=1, user descriptor
  g0[1] = lds_off;                              // lds_addr (bytes)
  g0[2] = (unsigned)(ga & 0xffffffffu);         // global_addr[31:0]
  g0[3] = (unsigned)((ga >> 32) & 0x1ffffffu)   // global_addr[56:32]
        | (2u << 30);                           // type = 2 ("image")
  i32x8 g1;
  g1[0] = (1 << 16)                             // data_size = 2 bytes
        | (1 << 20)                             // pad_enable
        | (3 << 22)                             // pad_interval: 16 DWORDs
        | (3 << 25);                            // pad_amount:   4 DWORDs
  g1[1] = (int)((1024u & 0xffffu) << 16);       // tensor_dim0 = 1024 (bits 79:48)
  g1[2] = (int)(((1024u >> 16) & 0xffffu)       // tensor_dim0 hi
        | ((1024u & 0xffffu) << 16));           // tensor_dim1 = 1024 (bits 111:80)
  g1[3] = (int)(((1024u >> 16) & 0xffffu)       // tensor_dim1 hi
        | (32u << 16));                         // tile_dim0 = 32 (bits 127:112)
  g1[4] = 128;                                  // tile_dim1 = 128, tile_dim2 = 0
  g1[5] = 1024;                                 // tensor_dim0_stride = 1024
  g1[6] = 0;
  g1[7] = 0;
  const i32x4 z4 = {0, 0, 0, 0};
  const i32x8 z8 = {0, 0, 0, 0, 0, 0, 0, 0};
  __builtin_amdgcn_tensor_load_to_lds(g0, g1, z4, z4, z8, 0);
}

// ---------------------------------------------------------------------------
// WMMA helper: D = A(16x32 f16) x B(32x16 f16) + C(16x16 f32)
// ---------------------------------------------------------------------------
static __device__ __forceinline__ v8f wmma16(v16h a, v16h b, v8f c) {
  return __builtin_amdgcn_wmma_f32_16x16x32_f16(
      /*neg_a=*/false, a, /*neg_b=*/false, b,
      /*c_mod=*/(short)0, c, /*reuse_a=*/false, /*reuse_b=*/false);
}

static __device__ __forceinline__ v16h cat8(v8h lo, v8h hi8) {
  v16h a;
#pragma unroll
  for (int i = 0; i < 8; ++i) { a[i] = lo[i]; a[i + 8] = hi8[i]; }
  return a;
}

static __device__ __forceinline__ v16h ld16_lds(const _Float16* p) {
  return cat8(*(const v8h*)p, *(const v8h*)(p + 8));
}

// A-fragment (16x32 f16) from an f32 row-major matrix (global).
static __device__ __forceinline__ v16h load_a_f32(const float* __restrict__ A, int lda,
                                                  int row0, int k0, int l, int hi) {
  const float* p = A + (size_t)(row0 + l) * lda + (k0 + hi * 16);
  v4f x0 = *(const v4f*)(p);
  v4f x1 = *(const v4f*)(p + 4);
  v4f x2 = *(const v4f*)(p + 8);
  v4f x3 = *(const v4f*)(p + 12);
  v16h a;
#pragma unroll
  for (int i = 0; i < 4; ++i) {
    a[i]      = (_Float16)x0[i];
    a[i + 4]  = (_Float16)x1[i];
    a[i + 8]  = (_Float16)x2[i];
    a[i + 12] = (_Float16)x3[i];
  }
  return a;
}

// A-fragment (16x32 f16) from an f16 row-major matrix (global).
static __device__ __forceinline__ v16h load_a_f16(const _Float16* __restrict__ A, int lda,
                                                  int row0, int k0, int l, int hi) {
  const _Float16* p = A + (size_t)(row0 + l) * lda + (k0 + hi * 16);
  return cat8(*(const v8h*)p, *(const v8h*)(p + 8));
}

// 16-lane-group reductions (C-fragment rows live across 16 lanes).
static __device__ __forceinline__ float redmax16(float v) {
  v = fmaxf(v, __shfl_xor(v, 1, 32));
  v = fmaxf(v, __shfl_xor(v, 2, 32));
  v = fmaxf(v, __shfl_xor(v, 4, 32));
  v = fmaxf(v, __shfl_xor(v, 8, 32));
  return v;
}
static __device__ __forceinline__ float redsum16(float v) {
  v += __shfl_xor(v, 1, 32);
  v += __shfl_xor(v, 2, 32);
  v += __shfl_xor(v, 4, 32);
  v += __shfl_xor(v, 8, 32);
  return v;
}

// ---------------------------------------------------------------------------
// 1) Weights f32 -> f16 (Wq | Wk | Wv | Wo packed consecutively)
// ---------------------------------------------------------------------------
__global__ void cvtw_kernel(const float* __restrict__ Wq, const float* __restrict__ Wk,
                            const float* __restrict__ Wv, const float* __restrict__ Wo,
                            _Float16* __restrict__ W16) {
  const int idx = blockIdx.x * blockDim.x + threadIdx.x;
  const float* src = (blockIdx.y == 0) ? Wq
                   : (blockIdx.y == 1) ? Wk
                   : (blockIdx.y == 2) ? Wv : Wo;
  W16[(size_t)blockIdx.y * D_ * D_ + idx] = (_Float16)src[idx];
}

// ---------------------------------------------------------------------------
// 2) QKV projection GEMM, async-LDS double-buffered weight tiles.
// ---------------------------------------------------------------------------
__global__ __launch_bounds__(256) void qkv_kernel(const float* __restrict__ q,
                                                  const float* __restrict__ k,
                                                  const float* __restrict__ v,
                                                  const _Float16* __restrict__ W16,
                                                  _Float16* __restrict__ Qo,
                                                  _Float16* __restrict__ Ko,
                                                  _Float16* __restrict__ Vto) {
  __shared__ _Float16 wtile[2][128 * 40];   // [col][k], padded stride 40 halves
  const int which = blockIdx.y;
  const float* A = (which == 0) ? q : (which == 1) ? k : v;
  const _Float16* W = W16 + (size_t)which * D_ * D_;
  const int tid  = threadIdx.x;
  const int w    = tid >> 5;
  const int lane = tid & 31;
  const int l  = lane & 15;
  const int hi = lane >> 4;
  const int row0 = blockIdx.x * 128 + w * 16;  // flat row in [B*S]
  const int n0   = blockIdx.z * 128;           // output-column block

  auto prefetch = [&](int k0, int buf) {
    const int c  = tid >> 1;
    const int m0 = (tid & 1) * 2;
#pragma unroll
    for (int j = 0; j < 2; ++j) {
      const int m = m0 + j;
      async_ld_b128((unsigned)(size_t)&wtile[buf][c * 40 + m * 8],
                    W + (size_t)(n0 + c) * D_ + k0 + m * 8);
    }
  };

  prefetch(0, 0);
  v8f acc[8] = {};
#pragma unroll 1
  for (int k0 = 0; k0 < D_; k0 += 32) {
    const int cur = (k0 >> 5) & 1;
    wait_async0();
    __syncthreads();
    if (k0 + 32 < D_) prefetch(k0 + 32, cur ^ 1);
    const v16h a = load_a_f32(A, D_, row0, k0, l, hi);
    // Batch all fragment loads first, then the WMMA chain (staggered waits).
    v16h bf[8];
#pragma unroll
    for (int t = 0; t < 8; ++t)
      bf[t] = ld16_lds(&wtile[cur][(t * 16 + l) * 40 + hi * 16]);
#pragma unroll
    for (int t = 0; t < 8; ++t)
      acc[t] = wmma16(a, bf[t], acc[t]);
  }

#pragma unroll
  for (int t = 0; t < 8; ++t) {
    const int d  = n0 + t * 16 + l;
    const int h  = d >> 6;
    const int hd = d & 63;
#pragma unroll
    for (int r = 0; r < 8; ++r) {
      const int row = row0 + r + hi * 8;
      const int b = row >> 11;
      const int s = row & (S_ - 1);
      const _Float16 val = (_Float16)acc[t][r];
      if (which == 2)
        Vto[(((size_t)b * H_ + h) * HD_ + hd) * S_ + s] = val;
      else {
        _Float16* dst = which ? Ko : Qo;
        dst[(((size_t)b * H_ + h) * S_ + s) * HD_ + hd] = val;
      }
    }
  }
}

// ---------------------------------------------------------------------------
// 3) RoPE (reference semantics: emb = concat(freqs,freqs); interleaved rotate)
// ---------------------------------------------------------------------------
__global__ void rope_kernel(_Float16* __restrict__ Q16, _Float16* __restrict__ K16) {
  const size_t idx  = (size_t)blockIdx.x * blockDim.x + threadIdx.x;  // B*H*S*32
  const int    i    = (int)(idx & 31);
  const size_t rest = idx >> 5;
  const int    s    = (int)(rest & (S_ - 1));
  const size_t bh   = rest >> 11;
  _Float16* p = (blockIdx.y ? K16 : Q16) + (bh * S_ + s) * HD_ + 2 * i;
  const float pos = (float)s;
  const int d0 = 2 * i, d1 = 2 * i + 1;
  const float f0 = __powf(10000.f, -(float)(d0 & 31) * (1.f / 32.f));
  const float f1 = __powf(10000.f, -(float)(d1 & 31) * (1.f / 32.f));
  const float t0 = pos * f0, t1 = pos * f1;
  const float x0 = (float)p[0], x1 = (float)p[1];
  p[0] = (_Float16)(x0 * __cosf(t0) - x1 * __sinf(t0));
  p[1] = (_Float16)(x1 * __cosf(t1) + x0 * __sinf(t1));
}

// ---------------------------------------------------------------------------
// 4) Flash attention with async-LDS double-buffered K/V tiles.
// ---------------------------------------------------------------------------
__global__ __launch_bounds__(256) void attn_kernel(const _Float16* __restrict__ Q,
                                                   const _Float16* __restrict__ K,
                                                   const _Float16* __restrict__ Vt,
                                                   _Float16* __restrict__ O) {
  __shared__ _Float16 ktile[2][32 * 72];   // [key][hd], padded stride 72
  __shared__ _Float16 vtile[2][64 * 40];   // [hd][key], padded stride 40
  __shared__ _Float16 pbuf[8][16 * 40];    // per-wave P tile (16x32)
  const int tid  = threadIdx.x;
  const int w    = tid >> 5;
  const int lane = tid & 31;
  const int l  = lane & 15;
  const int hi = lane >> 4;
  const int bh = blockIdx.y;
  const int b  = bh >> 4;
  const int h  = bh & 15;
  const _Float16* Qh = Q  + (size_t)bh * S_ * HD_;
  const _Float16* Kh = K  + (size_t)bh * S_ * HD_;
  const _Float16* Vh = Vt + (size_t)bh * HD_ * S_;
  const int q0 = blockIdx.x * 128 + w * 16;
  _Float16* pl = pbuf[w];

  auto prefetch = [&](int kt, int buf) {
    {  // K tile: 32 rows x 64 halves = 4KB -> 1 b128 per thread
      const int row = tid >> 3, m = tid & 7;
      async_ld_b128((unsigned)(size_t)&ktile[buf][row * 72 + m * 8],
                    Kh + (size_t)(kt + row) * HD_ + m * 8);
    }
    {  // Vt tile: 64 rows x 32 halves = 4KB -> 1 b128 per thread
      const int row = tid >> 2, m = tid & 3;
      async_ld_b128((unsigned)(size_t)&vtile[buf][row * 40 + m * 8],
                    Vh + (size_t)row * S_ + kt + m * 8);
    }
  };

  const v16h qa0 = load_a_f16(Qh, HD_, q0, 0,  l, hi);
  const v16h qa1 = load_a_f16(Qh, HD_, q0, 32, l, hi);

  v8f o0 = {}, o1 = {}, o2 = {}, o3 = {};
  float mrow[8], lrow[8];
#pragma unroll
  for (int r = 0; r < 8; ++r) { mrow[r] = -1e30f; lrow[r] = 0.f; }

  prefetch(0, 0);
#pragma unroll 1
  for (int kt = 0; kt < S_; kt += 32) {
    const int cur = (kt >> 5) & 1;
    wait_async0();
    __syncthreads();
    if (kt + 32 < S_) prefetch(kt + 32, cur ^ 1);

    // Batch K fragments, then score WMMAs.
    v16h kb[4];
    kb[0] = ld16_lds(&ktile[cur][(l)      * 72 + 0  + hi * 16]);
    kb[1] = ld16_lds(&ktile[cur][(l)      * 72 + 32 + hi * 16]);
    kb[2] = ld16_lds(&ktile[cur][(16 + l) * 72 + 0  + hi * 16]);
    kb[3] = ld16_lds(&ktile[cur][(16 + l) * 72 + 32 + hi * 16]);
    // Hoist V fragments: their LDS latency hides under the softmax VALU block.
    v16h vb[4];
    vb[0] = ld16_lds(&vtile[cur][(l)      * 40 + hi * 16]);
    vb[1] = ld16_lds(&vtile[cur][(16 + l) * 40 + hi * 16]);
    vb[2] = ld16_lds(&vtile[cur][(32 + l) * 40 + hi * 16]);
    vb[3] = ld16_lds(&vtile[cur][(48 + l) * 40 + hi * 16]);

    v8f s0 = {}, s1 = {};
    s0 = wmma16(qa0, kb[0], s0);
    s0 = wmma16(qa1, kb[1], s0);
    s1 = wmma16(qa0, kb[2], s1);
    s1 = wmma16(qa1, kb[3], s1);

#pragma unroll
    for (int r = 0; r < 8; ++r) {
      const float x0 = s0[r] * 0.125f;       // 1/sqrt(64)
      const float x1 = s1[r] * 0.125f;
      const float rowmax = redmax16(fmaxf(x0, x1));
      const float mnew  = fmaxf(mrow[r], rowmax);
      const float alpha = __expf(mrow[r] - mnew);
      const float p0 = __expf(x0 - mnew);
      const float p1 = __expf(x1 - mnew);
      lrow[r] = lrow[r] * alpha + redsum16(p0 + p1);
      mrow[r] = mnew;
      o0[r] *= alpha; o1[r] *= alpha; o2[r] *= alpha; o3[r] *= alpha;
      const int rr = r + hi * 8;             // C layout: row = vgpr + hi*8
      pl[rr * 40 + l]      = (_Float16)p0;   // keys kt..kt+15
      pl[rr * 40 + 16 + l] = (_Float16)p1;   // keys kt+16..kt+31
    }
    // Same-wave LDS RAW: wait for our ds_stores before re-reading as A-frag.
    asm volatile("s_wait_dscnt 0" ::: "memory");
    const _Float16* pp = pl + l * 40 + hi * 16;
    const v16h pa = cat8(*(const v8h*)pp, *(const v8h*)(pp + 8));

    o0 = wmma16(pa, vb[0], o0);
    o1 = wmma16(pa, vb[1], o1);
    o2 = wmma16(pa, vb[2], o2);
    o3 = wmma16(pa, vb[3], o3);
  }

#pragma unroll
  for (int r = 0; r < 8; ++r) {
    const float inv = 1.f / lrow[r];
    const int s = q0 + r + hi * 8;
    _Float16* dst = O + ((size_t)b * S_ + s) * D_ + h * HD_ + l;
    dst[0]  = (_Float16)(o0[r] * inv);
    dst[16] = (_Float16)(o1[r] * inv);
    dst[32] = (_Float16)(o2[r] * inv);
    dst[48] = (_Float16)(o3[r] * inv);
  }
}

// ---------------------------------------------------------------------------
// 5) Output projection via TDM-staged weight tiles:
//    wave 0 issues one tensor_load_to_lds 2D descriptor per k-step
//    (128x32-half tile, LDS-padded to 40-half rows), TENSORcnt-tracked.
// ---------------------------------------------------------------------------
__global__ __launch_bounds__(256) void oproj_kernel(const _Float16* __restrict__ Oa,
                                                    const _Float16* __restrict__ Wo,
                                                    float* __restrict__ out) {
  __shared__ _Float16 wtile[2][128 * 40];
  const int tid  = threadIdx.x;
  const int w    = tid >> 5;
  const int lane = tid & 31;
  const int l  = lane & 15;
  const int hi = lane >> 4;
  const int row0 = blockIdx.x * 128 + w * 16;
  const int n0   = blockIdx.z * 128;

  auto tdm_prefetch = [&](int k0, int buf) {
    tdm_load_tile_128x32((unsigned)(size_t)&wtile[buf][0],
                         Wo + (size_t)n0 * D_ + k0);
  };

  if (w == 0) tdm_prefetch(0, 0);
  v8f acc[8] = {};
#pragma unroll 1
  for (int k0 = 0; k0 < D_; k0 += 32) {
    const int cur = (k0 >> 5) & 1;
    if (w == 0) __builtin_amdgcn_s_wait_tensorcnt(0);
    __syncthreads();                      // tile visible to all waves
    if (w == 0 && k0 + 32 < D_) tdm_prefetch(k0 + 32, cur ^ 1);
    const v16h a = load_a_f16(Oa, D_, row0, k0, l, hi);
    v16h bf[8];
#pragma unroll
    for (int t = 0; t < 8; ++t)
      bf[t] = ld16_lds(&wtile[cur][(t * 16 + l) * 40 + hi * 16]);
#pragma unroll
    for (int t = 0; t < 8; ++t)
      acc[t] = wmma16(a, bf[t], acc[t]);
  }
#pragma unroll
  for (int t = 0; t < 8; ++t) {
#pragma unroll
    for (int r = 0; r < 8; ++r) {
      const int row = row0 + r + hi * 8;
      out[(size_t)row * D_ + n0 + t * 16 + l] = acc[t][r];
    }
  }
}

// ---------------------------------------------------------------------------
// Host-side launcher.  Workspace layout (bytes):
//   [0,8MB) W16 | [8,24MB) Q16 [B,H,S,64] | [24,40MB) K16 | [40,56MB) Vt16
//   [56,72MB) O16 [B,S,D]
// ---------------------------------------------------------------------------
extern "C" void kernel_launch(void* const* d_in, const int* in_sizes, int n_in,
                              void* d_out, int out_size, void* d_ws, size_t ws_size,
                              hipStream_t stream) {
  (void)in_sizes; (void)n_in; (void)out_size; (void)ws_size;
  const float* query = (const float*)d_in[0];
  const float* key   = (const float*)d_in[1];
  const float* value = (const float*)d_in[2];
  const float* Wq = (const float*)d_in[3];
  const float* Wk = (const float*)d_in[4];
  const float* Wv = (const float*)d_in[5];
  const float* Wo = (const float*)d_in[6];
  float* out = (float*)d_out;
  char* ws = (char*)d_ws;
  const size_t MB = 1024 * 1024;
  _Float16* W16  = (_Float16*)(ws);
  _Float16* Q16  = (_Float16*)(ws + 8 * MB);
  _Float16* K16  = (_Float16*)(ws + 24 * MB);
  _Float16* Vt16 = (_Float16*)(ws + 40 * MB);
  _Float16* O16  = (_Float16*)(ws + 56 * MB);

  cvtw_kernel<<<dim3(D_ * D_ / 256, 4), 256, 0, stream>>>(Wq, Wk, Wv, Wo, W16);
  qkv_kernel<<<dim3((B_ * S_) / 128, 3, D_ / 128), 256, 0, stream>>>(
      query, key, value, W16, Q16, K16, Vt16);
  rope_kernel<<<dim3((B_ * H_ * S_ * 32) / 256, 2), 256, 0, stream>>>(Q16, K16);
  attn_kernel<<<dim3(S_ / 128, B_ * H_), 256, 0, stream>>>(Q16, K16, Vt16, O16);
  oproj_kernel<<<dim3((B_ * S_) / 128, 1, D_ / 128), 256, 0, stream>>>(
      O16, W16 + (size_t)3 * D_ * D_, out);
}
